// LSTM_10471130268438
// MI455X (gfx1250) — matrix operand order repaired and checked by reading, not compiled, not run on hardware
//
#include <hip/hip_runtime.h>

// ---------------------------------------------------------------------------
// LSTM forward on MI455X (gfx1250, wave32, WMMA).
//
// Strategy:
//  * All GEMMs on v_wmma_f32_16x16x32_bf16 with fp32 accumulation.
//  * fp32 accuracy recovered via split-bf16: A = Ahi + Alo, B = Bhi + Blo,
//    A*B ~= Ahi*Bhi + Ahi*Blo + Alo*Bhi  (3 WMMAs per K-step of 32).
//  * Input projection xg = X @ Wih^T + b  precomputed as one big GEMM
//    (keeps the 512-step serial chain at K=1024/step). Falls back to a
//    fused K=2048 step if the workspace can't hold xg (512 MiB fp32).
//  * Weights (17 MB bf16) stay L2-resident across the 512 step kernels.
// ---------------------------------------------------------------------------

typedef __attribute__((ext_vector_type(16))) __bf16 bf16x16;
typedef __attribute__((ext_vector_type(8)))  __bf16 bf16x8;
typedef __attribute__((ext_vector_type(8)))  float  f32x8;

union BF16V { bf16x16 v; bf16x8 h[2]; };

// A-matrix 16x32 bf16 fragment (ISA 7.12.2): lane = L + 16*kh, L = row M.
// lane holds K = kh*8 + [0..7] and K = 16 + kh*8 + [0..7].
__device__ __forceinline__ bf16x16 load_a_frag(const __bf16* row, int k0, int kh) {
  BF16V u;
  const __bf16* p = row + k0 + kh * 8;
  u.h[0] = *reinterpret_cast<const bf16x8*>(p);
  u.h[1] = *reinterpret_cast<const bf16x8*>(p + 16);
  return u.v;
}

// B-matrix 32x16 bf16 fragment: lane L (0..15) = column N, lane half kh holds
// K = kh*8 + [0..7] and K = 16 + kh*8 + [0..7] (same as A).
__device__ __forceinline__ bf16x16 load_b_frag(const __bf16* row, int k0, int kh) {
  BF16V u;
  const __bf16* p = row + k0 + kh * 8;
  u.h[0] = *reinterpret_cast<const bf16x8*>(p);
  u.h[1] = *reinterpret_cast<const bf16x8*>(p + 16);
  return u.v;
}

#define WMMA_BF16(a, b, c) \
  __builtin_amdgcn_wmma_f32_16x16x32_bf16(false, (a), false, (b), (short)0, (c), false, false)

__device__ __forceinline__ float fast_sigmoid(float x) {
  return 1.0f / (1.0f + __expf(-x));
}
__device__ __forceinline__ float fast_tanh(float x) {
  return 1.0f - 2.0f / (__expf(2.0f * x) + 1.0f);
}

// ---------------------------------------------------------------------------
// fp32 -> (hi, lo) bf16 split
// ---------------------------------------------------------------------------
__global__ void split_f32_bf16(const float* __restrict__ src,
                               __bf16* __restrict__ hi, __bf16* __restrict__ lo,
                               long n) {
  long i = (long)blockIdx.x * blockDim.x + threadIdx.x;
  long stride = (long)gridDim.x * blockDim.x;
  for (; i < n; i += stride) {
    float v = src[i];
    __bf16 h = (__bf16)v;
    hi[i] = h;
    lo[i] = (__bf16)(v - (float)h);
  }
}

// zero h0 (hi/lo) and c0
__global__ void init_state(__bf16* __restrict__ h_hi, __bf16* __restrict__ h_lo,
                           float* __restrict__ c, long n) {
  long i = (long)blockIdx.x * blockDim.x + threadIdx.x;
  long stride = (long)gridDim.x * blockDim.x;
  for (; i < n; i += stride) {
    h_hi[i] = (__bf16)0.0f;
    h_lo[i] = (__bf16)0.0f;
    c[i] = 0.0f;
  }
}

// ---------------------------------------------------------------------------
// Input projection: xg[32768][4096] = Xbf @ Wih^T + (bias_ih + bias_hh)
// One wave computes a 16(M) x 64(N) strip (4 accumulators, A-frag reuse).
// Grid: 2048 M-tiles x 64 N-groups = 131072 wave tiles / 8 waves = 16384 WGs.
// ---------------------------------------------------------------------------
__global__ void __launch_bounds__(256)
xproj_gemm(const __bf16* __restrict__ xh, const __bf16* __restrict__ xl,
           const __bf16* __restrict__ wh, const __bf16* __restrict__ wl,
           const float* __restrict__ bih, const float* __restrict__ bhh,
           float* __restrict__ xg) {
  const int lane = threadIdx.x & 31;
  const int L = lane & 15;
  const int kh = lane >> 4;
  const int wt = blockIdx.x * 8 + (threadIdx.x >> 5);
  const int mt = wt & 2047;   // 2048 M tiles of 16
  const int nt = wt >> 11;    // 64 N groups of 64
  const long m0 = (long)mt * 16;
  const int n0 = nt * 64;

  const __bf16* aH = xh + (m0 + L) * 1024;
  const __bf16* aL = xl + (m0 + L) * 1024;
  const __bf16* bH = wh + (long)(n0 + L) * 1024;
  const __bf16* bL = wl + (long)(n0 + L) * 1024;

  f32x8 acc[4];
  acc[0] = {}; acc[1] = {}; acc[2] = {}; acc[3] = {};

  for (int k0 = 0; k0 < 1024; k0 += 32) {
    bf16x16 ah = load_a_frag(aH, k0, kh);
    bf16x16 al = load_a_frag(aL, k0, kh);
#pragma unroll
    for (int j = 0; j < 4; ++j) {
      const long boff = (long)j * 16 * 1024;
      bf16x16 bh = load_b_frag(bH + boff, k0, kh);
      bf16x16 bl = load_b_frag(bL + boff, k0, kh);
      acc[j] = WMMA_BF16(ah, bh, acc[j]);
      acc[j] = WMMA_BF16(ah, bl, acc[j]);
      acc[j] = WMMA_BF16(al, bh, acc[j]);
    }
  }

#pragma unroll
  for (int j = 0; j < 4; ++j) {
    const int col = n0 + j * 16 + L;
    const float bs = bih[col] + bhh[col];
#pragma unroll
    for (int v = 0; v < 8; ++v) {
      const long m = m0 + v + 8 * kh;
      xg[m * 4096 + col] = acc[j][v] + bs;
    }
  }
}

// ---------------------------------------------------------------------------
// One LSTM timestep.
// Grid: 256 WGs (4 M-tiles x 64 hidden-column tiles) x 128 threads (4 waves).
// Wave g computes gate g's 16x16 tile: gates[m0:m0+16, g*1024+j0 : +16]
//   = (usePre ? xg_t : x_t @ Wih^T + bias) + h @ Whh^T
// then all threads apply the element-wise LSTM cell update.
// h is double-buffered across steps (hin != hout) to avoid cross-WG hazards.
// ---------------------------------------------------------------------------
__global__ void __launch_bounds__(128)
lstm_step(const __bf16* __restrict__ xh, const __bf16* __restrict__ xl,
          const __bf16* __restrict__ wih_h, const __bf16* __restrict__ wih_l,
          const __bf16* __restrict__ whh_h, const __bf16* __restrict__ whh_l,
          const __bf16* __restrict__ hin_h, const __bf16* __restrict__ hin_l,
          __bf16* __restrict__ hout_h, __bf16* __restrict__ hout_l,
          float* __restrict__ c,
          const float* __restrict__ xg,
          const float* __restrict__ bih, const float* __restrict__ bhh,
          float* __restrict__ out, int t, int usePre) {
  __shared__ float g_lds[4][16][17];

  const int lane = threadIdx.x & 31;
  const int g = threadIdx.x >> 5;     // gate: 0=i, 1=f, 2=g, 3=o
  const int L = lane & 15;
  const int kh = lane >> 4;
  const int mt = blockIdx.x & 3;
  const int jt = blockIdx.x >> 2;
  const int m0 = mt * 16;
  const int j0 = jt * 16;
  const int n0 = g * 1024 + j0;       // column in 4096-wide gate space

  f32x8 acc = {};

  // ---- recurrent segment: h @ Whh^T (K = 1024) ----
  {
    const __bf16* aH = hin_h + (long)(m0 + L) * 1024;
    const __bf16* aL = hin_l + (long)(m0 + L) * 1024;
    const __bf16* bH = whh_h + (long)(n0 + L) * 1024;
    const __bf16* bL = whh_l + (long)(n0 + L) * 1024;
    for (int k0 = 0; k0 < 1024; k0 += 32) {
      bf16x16 ah = load_a_frag(aH, k0, kh);
      bf16x16 al = load_a_frag(aL, k0, kh);
      bf16x16 bh = load_b_frag(bH, k0, kh);
      bf16x16 bl = load_b_frag(bL, k0, kh);
      acc = WMMA_BF16(ah, bh, acc);
      acc = WMMA_BF16(ah, bl, acc);
      acc = WMMA_BF16(al, bh, acc);
    }
  }

  // ---- fused input segment: x_t @ Wih^T (only when xg not precomputed) ----
  if (!usePre) {
    const __bf16* aH = xh + ((long)(m0 + L) * 512 + t) * 1024;
    const __bf16* aL = xl + ((long)(m0 + L) * 512 + t) * 1024;
    const __bf16* bH = wih_h + (long)(n0 + L) * 1024;
    const __bf16* bL = wih_l + (long)(n0 + L) * 1024;
    for (int k0 = 0; k0 < 1024; k0 += 32) {
      bf16x16 ah = load_a_frag(aH, k0, kh);
      bf16x16 al = load_a_frag(aL, k0, kh);
      bf16x16 bh = load_b_frag(bH, k0, kh);
      bf16x16 bl = load_b_frag(bL, k0, kh);
      acc = WMMA_BF16(ah, bh, acc);
      acc = WMMA_BF16(ah, bl, acc);
      acc = WMMA_BF16(al, bh, acc);
    }
  }

  // ---- epilogue: complete gate pre-activations into LDS ----
  {
    const int col = n0 + L;
    const float bs = usePre ? 0.0f : (bih[col] + bhh[col]);
#pragma unroll
    for (int v = 0; v < 8; ++v) {
      const int ml = v + 8 * kh;
      float val = acc[v] + bs;
      if (usePre)
        val += xg[((long)(m0 + ml) * 512 + t) * 4096 + col];
      g_lds[g][ml][L] = val;
    }
  }
  __syncthreads();

  // ---- element-wise LSTM cell update (256 elements / 128 threads) ----
  for (int e = threadIdx.x; e < 256; e += 128) {
    const int ml = e >> 4;
    const int jl = e & 15;
    const float gi = fast_sigmoid(g_lds[0][ml][jl]);
    const float gf = fast_sigmoid(g_lds[1][ml][jl]);
    const float gg = fast_tanh(g_lds[2][ml][jl]);
    const float go = fast_sigmoid(g_lds[3][ml][jl]);
    const int idx = (m0 + ml) * 1024 + j0 + jl;
    const float cn = gf * c[idx] + gi * gg;
    c[idx] = cn;
    const float hv = go * fast_tanh(cn);
    out[((long)(m0 + ml) * 512 + t) * 1024 + j0 + jl] = hv;
    const __bf16 hb = (__bf16)hv;
    hout_h[idx] = hb;
    hout_l[idx] = (__bf16)(hv - (float)hb);
  }
}

// ---------------------------------------------------------------------------
// Host launcher
// ---------------------------------------------------------------------------
extern "C" void kernel_launch(void* const* d_in, const int* in_sizes, int n_in,
                              void* d_out, int out_size, void* d_ws, size_t ws_size,
                              hipStream_t stream) {
  const float* x   = (const float*)d_in[0];  // (64, 512, 1024)
  const float* wih = (const float*)d_in[1];  // (4096, 1024)
  const float* whh = (const float*)d_in[2];  // (4096, 1024)
  const float* bih = (const float*)d_in[3];  // (4096,)
  const float* bhh = (const float*)d_in[4];  // (4096,)
  float* out = (float*)d_out;                // (64, 512, 1024)
  (void)in_sizes; (void)n_in; (void)out_size;

  const long NB = 64, T = 512, D = 1024, H = 1024, G = 4096;
  const long xN = NB * T * D;   // 33,554,432
  const long wN = G * D;        // 4,194,304
  const long sN = NB * H;       // 65,536

  char* base = (char*)d_ws;
  size_t used = 0;
  auto alloc = [&](size_t bytes) -> void* {
    bytes = (bytes + 255) & ~(size_t)255;
    void* r = base + used;
    used += bytes;
    return r;
  };

  __bf16* wih_h = (__bf16*)alloc(wN * 2);
  __bf16* wih_l = (__bf16*)alloc(wN * 2);
  __bf16* whh_h = (__bf16*)alloc(wN * 2);
  __bf16* whh_l = (__bf16*)alloc(wN * 2);
  __bf16* x_h   = (__bf16*)alloc(xN * 2);
  __bf16* x_l   = (__bf16*)alloc(xN * 2);
  __bf16* hA_h  = (__bf16*)alloc(sN * 2);
  __bf16* hA_l  = (__bf16*)alloc(sN * 2);
  __bf16* hB_h  = (__bf16*)alloc(sN * 2);
  __bf16* hB_l  = (__bf16*)alloc(sN * 2);
  float*  c     = (float*) alloc(sN * 4);

  const size_t xgBytes = (size_t)NB * T * G * sizeof(float); // 512 MiB
  const int usePre = (used + xgBytes + 512 <= ws_size) ? 1 : 0;
  float* xg = usePre ? (float*)alloc(xgBytes) : nullptr;

  // Precision splits (deterministic, recomputed every call).
  split_f32_bf16<<<8192, 256, 0, stream>>>(wih, wih_h, wih_l, wN);
  split_f32_bf16<<<8192, 256, 0, stream>>>(whh, whh_h, whh_l, wN);
  split_f32_bf16<<<32768, 256, 0, stream>>>(x, x_h, x_l, xN);
  init_state<<<256, 256, 0, stream>>>(hA_h, hA_l, c, sN);

  if (usePre) {
    // 32768x4096x1024 GEMM + bias, fully parallel, all on WMMA.
    xproj_gemm<<<16384, 256, 0, stream>>>(x_h, x_l, wih_h, wih_l, bih, bhh, xg);
  }

  // Sequential recurrence: 512 step kernels, h double-buffered.
  for (int t = 0; t < 512; ++t) {
    const __bf16* hin_h = (t & 1) ? hB_h : hA_h;
    const __bf16* hin_l = (t & 1) ? hB_l : hA_l;
    __bf16* hout_h = (t & 1) ? hA_h : hB_h;
    __bf16* hout_l = (t & 1) ? hA_l : hB_l;
    lstm_step<<<256, 128, 0, stream>>>(x_h, x_l, wih_h, wih_l, whh_h, whh_l,
                                       hin_h, hin_l, hout_h, hout_l, c, xg,
                                       bih, bhh, out, t, usePre);
  }
}
